// LSTMCell_61959198212449
// MI455X (gfx1250) — matrix operand order repaired and checked by reading, not compiled, not run on hardware
//
#include <hip/hip_runtime.h>

typedef __attribute__((ext_vector_type(16))) _Float16 v16h;
typedef __attribute__((ext_vector_type(8)))  _Float16 v8h;
typedef __attribute__((ext_vector_type(8)))  float    v8f;
typedef __attribute__((ext_vector_type(4)))  float    v4f;

#define BDIM 16384
#define IDIM 512
#define HDIM 512
#define KDIM 1024            // concat(x,h) reduction length
#define NK   (KDIM / 32)     // 32 K-steps of 32

// Block tile: 64 (M) x 64 (N per gate) x 4 gates, 256 threads = 8 wave32.
#define A_HALFS 2048         // 64 rows * 32 k        (fragment-permuted)
#define B_HALFS 8192         // 4 gates * 32 k * 64 n (fragment-permuted)
#define BUF_HALFS (A_HALFS + B_HALFS)

__device__ __forceinline__ v8h cvt8(v4f a, v4f b) {
    v8h r;
    r[0] = (_Float16)a[0]; r[1] = (_Float16)a[1];
    r[2] = (_Float16)a[2]; r[3] = (_Float16)a[3];
    r[4] = (_Float16)b[0]; r[5] = (_Float16)b[1];
    r[6] = (_Float16)b[2]; r[7] = (_Float16)b[3];
    return r;
}

__device__ __forceinline__ float sigmoid_f(float v) {
    return 1.0f / (1.0f + __expf(-v));
}

__global__ __launch_bounds__(256)
void lstm_cell_wmma(const float* __restrict__ x,
                    const float* __restrict__ h,
                    const float* __restrict__ c,
                    const float* __restrict__ wih,
                    const float* __restrict__ whh,
                    const float* __restrict__ bih,
                    const float* __restrict__ bhh,
                    float* __restrict__ out)
{
    __shared__ __align__(32) _Float16 lds[2][BUF_HALFS];

    const int t     = threadIdx.x;
    const int lane  = t & 31;
    const int wave  = t >> 5;
    const int mt    = wave & 3;   // this wave's 16-row subtile (0..3)
    const int npair = wave >> 2;  // 0/1 -> n-subtiles {0,1} / {2,3}

    const int nBase = blockIdx.x * 64;   // column block within H
    const int mBase = blockIdx.y * 64;   // row block within B

    // ---------------- A staging geometry (per thread) ----------------
    // 2048 elements / 256 threads = 8 contiguous f32 each (one row chunk).
    const int aml = t >> 2;              // local row 0..63
    const int akb = (t & 3) << 3;        // k offset in step: 0,8,16,24
    // 16-bit A 16x32 fragment permutation (ISA 7.12.2):
    //   lane = r + ((k>>3)&1)*16 ; half = (k&7) + ((k>>4)&1)*8
    const int aLane   = (aml & 15) + (((akb >> 3) & 1) << 4);
    const int aHalf   = ((akb >> 4) & 1) << 3;      // 0 or 8
    const int aLdsIdx = (aml >> 4) * 512 + aLane * 16 + aHalf;
    const float* aRowX = x + (size_t)(mBase + aml) * IDIM;
    const float* aRowH = h + (size_t)(mBase + aml) * IDIM;

    // ---------------- B staging geometry (per thread) ----------------
    // Column-gather transpose: each thread owns 4 columns x 8 K-rows of one
    // gate's W tile.  Per K-row, 16 threads read 256 contiguous bytes
    // (coalesced b128); LDS stores become 4 aligned b128 writes.
    const int bq  = t & 15;              // n-quad: columns bq*4 .. bq*4+3
    const int brh = (t >> 4) & 1;        // row half within 16-k group
    const int bci = t >> 5;              // 0..7
    const int bg  = bci >> 1;            // gate 0..3
    const int bkh = bci & 1;             // k half: +0 or +16
    const int bColBase = nBase + (bq << 2);
    // 16-bit B 32x16 fragment layout: lane = n + ((k>>3)&1)*16 ; half = (k&7) + ((k>>4)&1)*8
    int bDst[4];
#pragma unroll
    for (int cidx = 0; cidx < 4; ++cidx) {
        const int nl    = (bq << 2) + cidx;          // local n 0..63
        const int laneB = (nl & 15) + (brh << 4);
        const int nt    = nl >> 4;
        bDst[cidx] = A_HALFS + ((bg * 4 + nt) * 32 + laneB) * 16 + bkh * 8;
    }

    v8h aReg;
    v8h bReg[4];   // one v8h per owned column (8 k-halfs each)

    auto load_stage = [&](int k0) {
        // ---- A: 8 contiguous f32 from one row of x|h ----
        const int ka = k0 + akb;
        const float* ap = (ka < IDIM) ? (aRowX + ka) : (aRowH + (ka - IDIM));
        v4f f0 = *(const v4f*)(ap);
        v4f f1 = *(const v4f*)(ap + 4);
        aReg = cvt8(f0, f1);

        // ---- B: 8 K-rows x 4 columns from w_ih|w_hh ----
        const int kRow0 = k0 + (bkh << 4) + (brh << 3);   // first k-row
        const float* wp = (kRow0 < IDIM)
            ? (wih + (size_t)(bg * IDIM + kRow0)          * HDIM + bColBase)
            : (whh + (size_t)(bg * IDIM + (kRow0 - IDIM)) * HDIM + bColBase);
#pragma unroll
        for (int kk = 0; kk < 8; ++kk) {
            v4f w = *(const v4f*)(wp + (size_t)kk * HDIM);
            bReg[0][kk] = (_Float16)w[0];
            bReg[1][kk] = (_Float16)w[1];
            bReg[2][kk] = (_Float16)w[2];
            bReg[3][kk] = (_Float16)w[3];
        }

        // speculative prefetch of next K-step's data (global_prefetch_b8)
        if (k0 + 32 < KDIM) {
            const int kan = ka + 32;
            const float* apn = (kan < IDIM) ? (aRowX + kan)
                                            : (aRowH + (kan - IDIM));
            __builtin_prefetch(apn, 0, 1);
            const int krn = kRow0 + 32;
            const float* wpn = (krn < IDIM)
                ? (wih + (size_t)(bg * IDIM + krn)          * HDIM + bColBase)
                : (whh + (size_t)(bg * IDIM + (krn - IDIM)) * HDIM + bColBase);
            __builtin_prefetch(wpn, 0, 1);
        }
    };

    auto store_stage = [&](int buf) {
        *(v8h*)&lds[buf][aLdsIdx] = aReg;          // 1 aligned 16B store
#pragma unroll
        for (int cidx = 0; cidx < 4; ++cidx) {     // 4 aligned 16B stores
            *(v8h*)&lds[buf][bDst[cidx]] = bReg[cidx];
        }
    };

    v8f acc[4][2] = {};   // [gate][n-subtile] accumulators, f32

    // prologue: fill buffer 0
    load_stage(0);
    store_stage(0);

    for (int ks = 0; ks < NK; ++ks) {
        __syncthreads();                       // buf(ks&1) staged & prev reads done
        const int buf = ks & 1;
        if (ks + 1 < NK) load_stage((ks + 1) * 32);   // global -> regs (overlaps WMMA)

        const v16h afrag = *(const v16h*)&lds[buf][mt * 512 + lane * 16];
#pragma unroll
        for (int j = 0; j < 2; ++j) {
            const int nt = npair * 2 + j;
#pragma unroll
            for (int g = 0; g < 4; ++g) {
                const v16h bfrag =
                    *(const v16h*)&lds[buf][A_HALFS + ((g * 4 + nt) * 32 + lane) * 16];
                acc[g][j] = __builtin_amdgcn_wmma_f32_16x16x32_f16(
                    false, afrag, false, bfrag, (short)0, acc[g][j], false, false);
            }
        }

        if (ks + 1 < NK) store_stage(buf ^ 1);  // regs -> LDS for next step
    }

    // ---------------- epilogue: gates + cell update ----------------
    // C/D layout: lanes 0-15 N=0..15 rows 0..7 ; lanes 16-31 N=0..15 rows 8..15
    const int nCol   = lane & 15;
    const int rowOff = (lane >> 4) * 8;
    const size_t BH  = (size_t)BDIM * HDIM;

#pragma unroll
    for (int j = 0; j < 2; ++j) {
        const int nt = npair * 2 + j;
        const int n  = nBase + nt * 16 + nCol;
        const float bi = bih[0 * HDIM + n] + bhh[0 * HDIM + n];
        const float bf = bih[1 * HDIM + n] + bhh[1 * HDIM + n];
        const float bc = bih[2 * HDIM + n] + bhh[2 * HDIM + n];
        const float bo = bih[3 * HDIM + n] + bhh[3 * HDIM + n];
#pragma unroll
        for (int v = 0; v < 8; ++v) {
            const int m = mBase + mt * 16 + rowOff + v;
            const size_t idx = (size_t)m * HDIM + n;
            const float ing = sigmoid_f(acc[0][j][v] + bi);
            const float fg  = sigmoid_f(acc[1][j][v] + bf);
            const float cg  = tanhf(acc[2][j][v] + bc);
            const float og  = sigmoid_f(acc[3][j][v] + bo);
            const float cnew = fg * c[idx] + ing * cg;
            const float hnew = og * tanhf(cnew);
            out[idx]          = hnew;   // state_h
            out[BH + idx]     = hnew;   // state_h (duplicated output)
            out[2 * BH + idx] = cnew;   // state_c
        }
    }
}

extern "C" void kernel_launch(void* const* d_in, const int* in_sizes, int n_in,
                              void* d_out, int out_size, void* d_ws, size_t ws_size,
                              hipStream_t stream) {
    (void)in_sizes; (void)n_in; (void)out_size; (void)d_ws; (void)ws_size;
    const float* x   = (const float*)d_in[0];
    const float* h   = (const float*)d_in[1];
    const float* c   = (const float*)d_in[2];
    const float* wih = (const float*)d_in[3];
    const float* whh = (const float*)d_in[4];
    const float* bih = (const float*)d_in[5];
    const float* bhh = (const float*)d_in[6];
    float* out = (float*)d_out;

    dim3 grid(HDIM / 64, BDIM / 64);   // 8 x 256 = 2048 workgroups
    lstm_cell_wmma<<<grid, 256, 0, stream>>>(x, h, c, wih, whh, bih, bhh, out);
}
